// MoeTokenLevelFeedForwardGshard_33466385170913
// MI455X (gfx1250) — compile-verified, matrix operand order checked by reading, not run on hardware
//
#include <hip/hip_runtime.h>
#include <stdint.h>

typedef __attribute__((ext_vector_type(8)))  __bf16 bf16x8;
typedef __attribute__((ext_vector_type(16))) __bf16 bf16x16;
typedef __attribute__((ext_vector_type(8)))  float  f32x8;
typedef __attribute__((ext_vector_type(4)))  unsigned int u32x4;
typedef __attribute__((ext_vector_type(8)))  int    i32x8;
typedef __attribute__((ext_vector_type(4)))  int    i32x4;

constexpr int NN  = 8192;
constexpr int DD  = 1024;
constexpr int DFF = 4096;
constexpr int EE  = 4;
constexpr int CAP = NN / EE;           // 2048
constexpr float EPSV = 2.2204460492503131e-16f;   // np.finfo(float).eps cast to f32

// --------------------------------------------- Tensor Data Mover support

#if __has_builtin(__builtin_amdgcn_tensor_load_to_lds) && __has_builtin(__builtin_amdgcn_s_wait_tensorcnt)
#define HAVE_TDM 1

// D# group 0 (ISA 8.3): count=1 | lds_addr | global_addr[56:0] | type=2
__device__ __forceinline__ u32x4 tdm_g0(unsigned lds_off, const void* gaddr) {
    uint64_t ga = (uint64_t)(uintptr_t)gaddr;
    u32x4 g;
    g[0] = 1u;                                   // count=1, user mode, no gather
    g[1] = lds_off;                              // lds_addr (bytes)
    g[2] = (unsigned)(ga & 0xffffffffu);         // global_addr[31:0]
    g[3] = (unsigned)((ga >> 32) & 0x1ffffffu)   // global_addr[56:32]
         | (2u << 30);                           // type = 2 ("image")
    return g;
}

// D# group 1 (ISA 8.4): bf16 2D tile with LDS row padding (+16B per 64B row)
__device__ __forceinline__ i32x8 tdm_g1(unsigned tensor_d0, unsigned tensor_d1,
                                        unsigned tile_d0, unsigned tile_d1,
                                        unsigned stride_d0) {
    i32x8 g;
    g[0] = (int)((1u << 16)       // data_size = 2 bytes
               | (1u << 20)       // pad_enable (LDS dest padding)
               | (3u << 22)       // pad_interval: 16 DWORDs (= one 64B tile row)
               | (3u << 25));     // pad_amount:   4 DWORDs (= 16B -> 40-elem rows)
    g[1] = (int)((tensor_d0 & 0xffffu) << 16);                 // tensor_dim0[15:0]
    g[2] = (int)(((tensor_d0 >> 16) & 0xffffu)
               | ((tensor_d1 & 0xffffu) << 16));               // d0 hi | d1 lo
    g[3] = (int)(((tensor_d1 >> 16) & 0xffffu)
               | ((tile_d0 & 0xffffu) << 16));                 // d1 hi | tile_dim0
    g[4] = (int)(tile_d1 & 0xffffu);                           // tile_dim1 (tile_dim2=0)
    g[5] = (int)stride_d0;                                     // tensor_dim0_stride lo32
    g[6] = 0;                                                  // stride hi / dim1_stride
    g[7] = 0;
    return g;
}

__device__ __forceinline__ void tdm_load(u32x4 g0, i32x8 g1) {
    i32x4 z = (i32x4){0, 0, 0, 0};
#if __clang_major__ >= 23
    i32x8 z8 = (i32x8){0, 0, 0, 0, 0, 0, 0, 0};
    __builtin_amdgcn_tensor_load_to_lds(g0, g1, z, z, z8, 0);
#else
    __builtin_amdgcn_tensor_load_to_lds(g0, g1, z, z, 0);
#endif
}
#endif

__device__ __forceinline__ void wait_tensor0() {
#if defined(HAVE_TDM)
    __builtin_amdgcn_s_wait_tensorcnt(0);
#endif
}
__device__ __forceinline__ void wait_tensor1() {
#if defined(HAVE_TDM)
    __builtin_amdgcn_s_wait_tensorcnt(1);
#endif
}
__device__ __forceinline__ void wait_tensor2() {
#if defined(HAVE_TDM)
    __builtin_amdgcn_s_wait_tensorcnt(2);
#endif
}

// ---------------------------------------------------------------- utilities

__device__ __forceinline__ bf16x16 load_frag(const __bf16* row, int hl) {
    // CDNA5 16-bit A/B fragment: VGPRs 0..3 hold K = hl*8 + 0..7 (packed pairs),
    // VGPRs 4..7 hold K = 16 + hl*8 + 0..7.  Two 16B LDS vector loads.
    bf16x8 lo = *reinterpret_cast<const bf16x8*>(row + hl * 8);
    bf16x8 hi = *reinterpret_cast<const bf16x8*>(row + 16 + hl * 8);
    bf16x16 r;
#pragma unroll
    for (int i = 0; i < 8; ++i) { r[i] = lo[i]; r[8 + i] = hi[i]; }
    return r;
}

// ------------------------------------------------------------- zero / setup

__global__ void zero_kernel(float* __restrict__ y, int* __restrict__ hdr) {
    size_t i = (size_t)blockIdx.x * blockDim.x + threadIdx.x;
    if (i < (size_t)NN * DD) y[i] = 0.0f;
    if (i < 12) hdr[i] = 0;               // Mcnt[4], gsum[4], cnt1[4]
}

// src: f32 [K][N] row-major  ->  dst: bf16 [N][K] (column-major of src)
__global__ void transpose_kernel(const float* __restrict__ src,
                                 __bf16* __restrict__ dst, int Kd, int Nd) {
    const size_t mat = (size_t)Kd * Nd;
    const float* s = src + (size_t)blockIdx.z * mat;
    __bf16*      d = dst + (size_t)blockIdx.z * mat;
    __shared__ float t[32][33];
    const int n0 = blockIdx.x * 32, k0 = blockIdx.y * 32;
    const int tx = threadIdx.x, ty = threadIdx.y;   // (32, 8)
#pragma unroll
    for (int j = 0; j < 4; ++j)
        t[ty + j * 8][tx] = s[(size_t)(k0 + ty + j * 8) * Nd + n0 + tx];
    __syncthreads();
#pragma unroll
    for (int j = 0; j < 4; ++j)
        d[(size_t)(n0 + ty + j * 8) * Kd + k0 + tx] = (__bf16)t[tx][ty + j * 8];
}

// ------------------------------------------------------------------ gating

__global__ __launch_bounds__(256) void gating_kernel(
    const float* __restrict__ x, const float* __restrict__ wg,
    const float* __restrict__ rand_u,
    float* __restrict__ cw, float* __restrict__ gsum, float* __restrict__ cnt1) {
    __shared__ float sg[4], sc[4];
    if (threadIdx.x < 4) { sg[threadIdx.x] = 0.f; sc[threadIdx.x] = 0.f; }
    __syncthreads();
    const int token = blockIdx.x * 8 + (threadIdx.x >> 5);   // one wave32 per token
    const int lane  = threadIdx.x & 31;
    const float* xr = x + (size_t)token * DD;
    float p0 = 0.f, p1 = 0.f, p2 = 0.f, p3 = 0.f;
    for (int k = lane; k < DD; k += 32) {
        float xv = xr[k];
        const float* wr = wg + k * 4;
        p0 += xv * wr[0]; p1 += xv * wr[1]; p2 += xv * wr[2]; p3 += xv * wr[3];
    }
#pragma unroll
    for (int off = 16; off; off >>= 1) {
        p0 += __shfl_xor(p0, off); p1 += __shfl_xor(p1, off);
        p2 += __shfl_xor(p2, off); p3 += __shfl_xor(p3, off);
    }
    if (lane == 0) {
        float m = fmaxf(fmaxf(p0, p1), fmaxf(p2, p3));
        float e0 = __expf(p0 - m), e1 = __expf(p1 - m), e2 = __expf(p2 - m), e3 = __expf(p3 - m);
        float s = e0 + e1 + e2 + e3;
        float g[4] = { e0 / s, e1 / s, e2 / s, e3 / s };
        // top-2, stable (lowest index first on ties) like lax.top_k
        float bv = -1.f; int bi = 0; float sv = -1.f; int si = 0;
#pragma unroll
        for (int e = 0; e < 4; ++e) {
            float v = g[e];
            if (v > bv) { sv = bv; si = bi; bv = v; bi = e; }
            else if (v > sv) { sv = v; si = e; }
        }
        float s2 = bv + sv;
        float t0 = bv / s2, t1 = sv / s2;
        float c2 = (t1 > rand_u[(size_t)token * 4 + si]) ? t1 : 0.f;
        float row[4] = { 0.f, 0.f, 0.f, 0.f };
        row[bi] = t0; row[si] = c2;
#pragma unroll
        for (int e = 0; e < 4; ++e) cw[(size_t)token * 4 + e] = row[e];
#pragma unroll
        for (int e = 0; e < 4; ++e) atomicAdd(&sg[e], g[e]);
        atomicAdd(&sc[bi], 1.f);
    }
    __syncthreads();
    if (threadIdx.x < 4) {
        atomicAdd(&gsum[threadIdx.x], sg[threadIdx.x]);
        atomicAdd(&cnt1[threadIdx.x], sc[threadIdx.x]);
    }
}

// ------------------------------------------------- capacity (top-2048/expert)

__global__ __launch_bounds__(1024) void capacity_kernel(float* __restrict__ cw) {
    const int e = blockIdx.x;
    const int tid = threadIdx.x;
    __shared__ float col[NN];
    __shared__ int   scn[1024];
    __shared__ int   s_cnt;
    for (int s = 0; s < NN / 1024; ++s)
        col[tid + s * 1024] = cw[(size_t)(tid + s * 1024) * 4 + e];
    __syncthreads();
    // values are >= 0: float bits are order-isomorphic -> binary search for the
    // CAP-th largest value's bit pattern.  T = min{ u : #{v > u} < CAP }.
    unsigned lo = 0u, hi = 0x7f800000u;
    while (lo < hi) {
        unsigned mid = lo + ((hi - lo) >> 1);
        float t = __uint_as_float(mid);
        int c = 0;
#pragma unroll
        for (int s = 0; s < 8; ++s) c += (col[tid + s * 1024] > t) ? 1 : 0;
#pragma unroll
        for (int off = 16; off; off >>= 1) c += __shfl_xor(c, off);
        if (tid == 0) s_cnt = 0;
        __syncthreads();
        if ((tid & 31) == 0) atomicAdd(&s_cnt, c);
        __syncthreads();
        int total = s_cnt;
        __syncthreads();
        if (total < CAP) hi = mid; else lo = mid + 1;
    }
    const float thr = __uint_as_float(lo);
    {   // count strictly greater
        int c = 0;
#pragma unroll
        for (int s = 0; s < 8; ++s) c += (col[tid + s * 1024] > thr) ? 1 : 0;
#pragma unroll
        for (int off = 16; off; off >>= 1) c += __shfl_xor(c, off);
        if (tid == 0) s_cnt = 0;
        __syncthreads();
        if ((tid & 31) == 0) atomicAdd(&s_cnt, c);
        __syncthreads();
    }
    const int remaining = CAP - s_cnt;
    __syncthreads();
    // tie ranks in global index order (stable top_k tie-break)
    const int base = tid * 8;
    int myT = 0;
#pragma unroll
    for (int j = 0; j < 8; ++j) myT += (col[base + j] == thr) ? 1 : 0;
    scn[tid] = myT;
    __syncthreads();
    for (int off = 1; off < 1024; off <<= 1) {
        int v = (tid >= off) ? scn[tid - off] : 0;
        __syncthreads();
        scn[tid] += v;
        __syncthreads();
    }
    int rank = scn[tid] - myT;
    const bool thrPos = thr > 0.f;
#pragma unroll
    for (int j = 0; j < 8; ++j) {
        float v = col[base + j];
        bool keep = (v > thr);
        if (v == thr) { if (thrPos && rank < remaining) keep = true; rank++; }
        cw[(size_t)(base + j) * 4 + e] = keep ? v : 0.f;
    }
}

// ------------------------------------- random re-route + compaction lists

__global__ void finalize_kernel(float* __restrict__ gates,
                                const float* __restrict__ rand_vals,
                                const int* __restrict__ rand_idx,
                                int* __restrict__ lists, int* __restrict__ Mcnt) {
    const int i = blockIdx.x * blockDim.x + threadIdx.x;
    if (i >= NN) return;
    float g[4]; bool any = false;
#pragma unroll
    for (int e = 0; e < 4; ++e) { g[e] = gates[(size_t)i * 4 + e]; any = any || (g[e] > 0.f); }
    if (!any) {
        float r0 = rand_vals[i * 2], r1 = rand_vals[i * 2 + 1];
        float s = r0 + r1; r0 /= s; r1 /= s;
#pragma unroll
        for (int e = 0; e < 4; ++e) g[e] = 0.f;
        g[rand_idx[i * 2]] = r0;
        g[rand_idx[i * 2 + 1]] = r1;           // last-write-wins like .at[].set
#pragma unroll
        for (int e = 0; e < 4; ++e) gates[(size_t)i * 4 + e] = g[e];
    }
#pragma unroll
    for (int e = 0; e < 4; ++e)
        if (g[e] > 0.f) {
            int p = atomicAdd(&Mcnt[e], 1);
            lists[e * NN + p] = i;
        }
}

// ----------------------------------- GEMM1: h = relu(gather(x) @ W1[e] + b1)

__global__ __launch_bounds__(256) void gemm1_kernel(
    const float* __restrict__ x, const __bf16* __restrict__ W1t,
    const float* __restrict__ b1, const int* __restrict__ lists,
    const int* __restrict__ Mcnt, __bf16* __restrict__ h, int e) {
    const int Me = Mcnt[e];
    const int m0 = blockIdx.y * 128;
    if (m0 >= Me) return;                       // uniform early exit (Me>=1 past here)
    const int n0 = blockIdx.x * 128;
    __shared__ __align__(16) __bf16 As[128][40];
    __shared__ __align__(16) __bf16 Bs[2][128][40];   // Bs[n][k], double buffered
    const int tid = threadIdx.x;
    const int wave = tid >> 5, lane = tid & 31;
    const int wm = wave >> 2, wn = wave & 3;          // 2 x 4 wave grid
    const int hl = lane >> 4, lm = lane & 15;
    const __bf16* Bt = W1t + (size_t)e * DFF * DD;    // [DFF][DD]
    const int* list = lists + e * NN;
    const int arow = tid >> 1, acol = (tid & 1) << 4;
    int rr = m0 + arow; if (rr > Me - 1) rr = Me - 1; // clamp: dead rows never stored
    const int tok_a = list[rr];
    const float* xrow = x + (size_t)tok_a * DD + acol;

    f32x8 acc[4][2];
#pragma unroll
    for (int mi = 0; mi < 4; ++mi)
#pragma unroll
        for (int ni = 0; ni < 2; ++ni)
            acc[mi][ni] = (f32x8){0.f,0.f,0.f,0.f,0.f,0.f,0.f,0.f};

#if defined(HAVE_TDM)
    const i32x8 g1B = tdm_g1(/*tensor_d0=*/32, /*tensor_d1=*/128,
                             /*tile_d0=*/32, /*tile_d1=*/128, /*stride=*/DD);
#endif
    auto stageB = [&](int buf, int kt) {
#if defined(HAVE_TDM)
        if (wave == 0) {
            unsigned lds = (unsigned)(uintptr_t)&Bs[buf][0][0];
            const __bf16* ga = Bt + (size_t)n0 * DD + (size_t)kt * 32;
            tdm_load(tdm_g0(lds, ga), g1B);
        }
#else
        const __bf16* g = Bt + (size_t)(n0 + arow) * DD + (size_t)kt * 32 + acol;
        __bf16* db = &Bs[buf][arow][acol];
        *reinterpret_cast<bf16x8*>(db)     = *reinterpret_cast<const bf16x8*>(g);
        *reinterpret_cast<bf16x8*>(db + 8) = *reinterpret_cast<const bf16x8*>(g + 8);
#endif
    };
    auto stageA = [&](int kt) {   // gathered f32 -> bf16 through VGPRs
        const float4* p = reinterpret_cast<const float4*>(xrow + kt * 32);
        __bf16* da = &As[arow][acol];
#pragma unroll
        for (int c = 0; c < 4; ++c) {
            float4 q = p[c];
            da[c * 4 + 0] = (__bf16)q.x; da[c * 4 + 1] = (__bf16)q.y;
            da[c * 4 + 2] = (__bf16)q.z; da[c * 4 + 3] = (__bf16)q.w;
        }
    };

    constexpr int KT = DD / 32;
    stageB(0, 0);
    for (int kt = 0; kt < KT; ++kt) {
        const int cur = kt & 1;
        stageA(kt);
        if (kt + 1 < KT) { stageB(cur ^ 1, kt + 1); if (wave == 0) wait_tensor1(); }
        else             { if (wave == 0) wait_tensor0(); }
        __syncthreads();
        bf16x16 af[4], bq[2];
#pragma unroll
        for (int mi = 0; mi < 4; ++mi) af[mi] = load_frag(&As[wm * 64 + mi * 16 + lm][0], hl);
#pragma unroll
        for (int ni = 0; ni < 2; ++ni) bq[ni] = load_frag(&Bs[cur][wn * 32 + ni * 16 + lm][0], hl);
#pragma unroll
        for (int mi = 0; mi < 4; ++mi)
#pragma unroll
            for (int ni = 0; ni < 2; ++ni)
                acc[mi][ni] = __builtin_amdgcn_wmma_f32_16x16x32_bf16(
                    false, af[mi], false, bq[ni], (short)0, acc[mi][ni], false, false);
        __syncthreads();
    }
#pragma unroll
    for (int mi = 0; mi < 4; ++mi)
#pragma unroll
        for (int ni = 0; ni < 2; ++ni) {
            const int ng = n0 + wn * 32 + ni * 16 + lm;
            const float bias = b1[(size_t)e * DFF + ng];
#pragma unroll
            for (int i = 0; i < 8; ++i) {
                const int mg = m0 + wm * 64 + mi * 16 + hl * 8 + i;  // C layout: M = hl*8+i
                if (mg < Me) {
                    float v = acc[mi][ni][i] + bias;
                    h[(size_t)mg * DFF + ng] = (__bf16)(v > 0.f ? v : 0.f);
                }
            }
        }
}

// ------------------------- GEMM2: y[tok] += gate * (h @ W2[e] + b2), scatter

__global__ __launch_bounds__(256) void gemm2_kernel(
    const __bf16* __restrict__ h, const __bf16* __restrict__ W2t,
    const float* __restrict__ b2, const float* __restrict__ gates,
    const int* __restrict__ lists, const int* __restrict__ Mcnt,
    float* __restrict__ y, int e) {
    const int Me = Mcnt[e];
    const int m0 = blockIdx.y * 128;
    if (m0 >= Me) return;
    const int n0 = blockIdx.x * 128;
    __shared__ __align__(16) __bf16 As[2][128][40];
    __shared__ __align__(16) __bf16 Bs[2][128][40];   // Bs[n][k]
    const int tid = threadIdx.x;
    const int wave = tid >> 5, lane = tid & 31;
    const int wm = wave >> 2, wn = wave & 3;
    const int hl = lane >> 4, lm = lane & 15;
    const __bf16* Bt = W2t + (size_t)e * DD * DFF;    // [DD][DFF]
    const int arow = tid >> 1, acol = (tid & 1) << 4;
    int rr = m0 + arow; if (rr > Me - 1) rr = Me - 1; // clamp (fallback path only)
    const __bf16* hrow = h + (size_t)rr * DFF + acol;

    f32x8 acc[4][2];
#pragma unroll
    for (int mi = 0; mi < 4; ++mi)
#pragma unroll
        for (int ni = 0; ni < 2; ++ni)
            acc[mi][ni] = (f32x8){0.f,0.f,0.f,0.f,0.f,0.f,0.f,0.f};

#if defined(HAVE_TDM)
    unsigned remRows = (unsigned)(Me - m0);           // TDM zero-fills OOB rows
    const i32x8 g1A = tdm_g1(32, remRows, 32, 128, DFF);
    const i32x8 g1B = tdm_g1(32, 128,    32, 128, DFF);
#endif
    auto stage = [&](int buf, int kt) {
#if defined(HAVE_TDM)
        if (wave == 0) {
            unsigned ldsA = (unsigned)(uintptr_t)&As[buf][0][0];
            unsigned ldsB = (unsigned)(uintptr_t)&Bs[buf][0][0];
            const __bf16* ga = h  + (size_t)m0 * DFF + (size_t)kt * 32;
            const __bf16* gb = Bt + (size_t)n0 * DFF + (size_t)kt * 32;
            tdm_load(tdm_g0(ldsA, ga), g1A);
            tdm_load(tdm_g0(ldsB, gb), g1B);
        }
#else
        const __bf16* ga = hrow + (size_t)kt * 32;
        const __bf16* gb = Bt + (size_t)(n0 + arow) * DFF + (size_t)kt * 32 + acol;
        __bf16* da = &As[buf][arow][acol];
        __bf16* db = &Bs[buf][arow][acol];
        *reinterpret_cast<bf16x8*>(da)     = *reinterpret_cast<const bf16x8*>(ga);
        *reinterpret_cast<bf16x8*>(da + 8) = *reinterpret_cast<const bf16x8*>(ga + 8);
        *reinterpret_cast<bf16x8*>(db)     = *reinterpret_cast<const bf16x8*>(gb);
        *reinterpret_cast<bf16x8*>(db + 8) = *reinterpret_cast<const bf16x8*>(gb + 8);
#endif
    };

    constexpr int KT = DFF / 32;
    stage(0, 0);
    for (int kt = 0; kt < KT; ++kt) {
        const int cur = kt & 1;
        if (kt + 1 < KT) { stage(cur ^ 1, kt + 1); if (wave == 0) wait_tensor2(); }
        else             { if (wave == 0) wait_tensor0(); }
        __syncthreads();
        bf16x16 af[4], bq[2];
#pragma unroll
        for (int mi = 0; mi < 4; ++mi) af[mi] = load_frag(&As[cur][wm * 64 + mi * 16 + lm][0], hl);
#pragma unroll
        for (int ni = 0; ni < 2; ++ni) bq[ni] = load_frag(&Bs[cur][wn * 32 + ni * 16 + lm][0], hl);
#pragma unroll
        for (int mi = 0; mi < 4; ++mi)
#pragma unroll
            for (int ni = 0; ni < 2; ++ni)
                acc[mi][ni] = __builtin_amdgcn_wmma_f32_16x16x32_bf16(
                    false, af[mi], false, bq[ni], (short)0, acc[mi][ni], false, false);
        __syncthreads();
    }
    const int* list = lists + e * NN;
#pragma unroll
    for (int mi = 0; mi < 4; ++mi)
#pragma unroll
        for (int i = 0; i < 8; ++i) {
            const int mg = m0 + wm * 64 + mi * 16 + hl * 8 + i;
            if (mg >= Me) continue;
            const int tok = list[mg];
            const float gate = gates[(size_t)tok * 4 + e];
            float* yr = y + (size_t)tok * DD;
#pragma unroll
            for (int ni = 0; ni < 2; ++ni) {
                const int ng = n0 + wn * 32 + ni * 16 + lm;
                atomicAdd(yr + ng, gate * (acc[mi][ni][i] + b2[(size_t)e * DD + ng]));
            }
        }
}

// -------------------------------------------------- epsilon + aux loss

__global__ void epsloss_kernel(float* __restrict__ out,
                               const float* __restrict__ gsum,
                               const float* __restrict__ cnt1) {
    const size_t total = (size_t)NN * DD;
    size_t i = (size_t)blockIdx.x * blockDim.x + threadIdx.x;
    if (i < total && out[i] == 0.f) out[i] = EPSV;
    if (i == 0) {
        float l = 0.f;
#pragma unroll
        for (int e = 0; e < 4; ++e)
            l += (cnt1[e] / (float)NN) * (gsum[e] / (float)NN);
        out[total] = 0.1f * (l / (float)EE);
    }
}

// ------------------------------------------------------------------ launch

extern "C" void kernel_launch(void* const* d_in, const int* in_sizes, int n_in,
                              void* d_out, int out_size, void* d_ws, size_t ws_size,
                              hipStream_t stream) {
    (void)in_sizes; (void)n_in; (void)out_size; (void)ws_size;
    const float* x     = (const float*)d_in[0];
    const float* wgate = (const float*)d_in[1];
    const float* W1    = (const float*)d_in[2];
    const float* b1    = (const float*)d_in[3];
    const float* W2    = (const float*)d_in[4];
    const float* b2    = (const float*)d_in[5];
    const float* ru    = (const float*)d_in[6];
    const float* rv    = (const float*)d_in[7];
    const int*   ri    = (const int*)d_in[8];
    float* y = (float*)d_out;

    char* w = (char*)d_ws;
    int*    Mcnt  = (int*)w;                                     // 4 ints
    float*  gsum  = (float*)(w + 16);                            // 4 floats
    float*  cnt1  = (float*)(w + 32);                            // 4 floats
    float*  gates = (float*)(w + 256);                           // N*E f32
    int*    lists = (int*)(w + 256 + (size_t)NN * EE * 4);       // N*E int
    __bf16* W1t   = (__bf16*)(w + 256 + 2 * (size_t)NN * EE * 4);
    __bf16* W2t   = W1t + (size_t)EE * DD * DFF;
    __bf16* hbuf  = W2t + (size_t)EE * DFF * DD;                 // N*DFF bf16

    const int nblk = (NN * DD + 255) / 256;
    zero_kernel<<<nblk, 256, 0, stream>>>(y, (int*)w);
    transpose_kernel<<<dim3(DFF / 32, DD / 32, EE), dim3(32, 8), 0, stream>>>(W1, W1t, DD, DFF);
    transpose_kernel<<<dim3(DD / 32, DFF / 32, EE), dim3(32, 8), 0, stream>>>(W2, W2t, DFF, DD);
    gating_kernel<<<NN / 8, 256, 0, stream>>>(x, wgate, ru, gates, gsum, cnt1);
    capacity_kernel<<<EE, 1024, 0, stream>>>(gates);
    finalize_kernel<<<(NN + 255) / 256, 256, 0, stream>>>(gates, rv, ri, lists, Mcnt);
    for (int e = 0; e < EE; ++e) {
        gemm1_kernel<<<dim3(DFF / 128, NN / 128), 256, 0, stream>>>(
            x, W1t, b1, lists, Mcnt, hbuf, e);
        gemm2_kernel<<<dim3(DD / 128, NN / 128), 256, 0, stream>>>(
            hbuf, W2t, b2, gates, lists, Mcnt, y, e);
    }
    epsloss_kernel<<<nblk, 256, 0, stream>>>(y, gsum, cnt1);
}
